// RecurrenceAttention_86114094285167
// MI455X (gfx1250) — matrix operand order repaired and checked by reading, not compiled, not run on hardware
//
#include <hip/hip_runtime.h>
#include <math.h>

// ---------------------------------------------------------------------------
// Transformer-XL recurrence attention for gfx1250 (MI455X).
// Strategy: cast to bf16 once, do every GEMM with v_wmma_f32_16x16x32_bf16
// (fp32 accumulation), fold rel_shift into the A-operand row gather of the
// scores GEMM, materialize bf16 attention scores (128MB, ~11us at 23.3TB/s)
// because the reference softmaxes over the QUERY axis (column softmax),
// which prevents single-pass flash-attention fusion.
// Workspace requirement: ~206 MB.
// ---------------------------------------------------------------------------

#define DM   1024
#define NH   16
#define DH   64
#define BB   2
#define CUR  1024
#define PREV 1024
#define TT   2048      // T = CUR + PREV
#define II   1024      // query length
#define JJ   2048      // key length

typedef __attribute__((ext_vector_type(16))) __bf16 bf16x16;
typedef __attribute__((ext_vector_type(8)))  float  floatx8;

// ----------------------------- small helpers -------------------------------

__device__ __forceinline__ unsigned short f2b(float f) {
    unsigned u = __float_as_uint(f);
    u += 0x7FFFu + ((u >> 16) & 1u);      // round-to-nearest-even; -inf safe
    return (unsigned short)(u >> 16);
}
__device__ __forceinline__ float b2f(unsigned short h) {
    return __uint_as_float(((unsigned)h) << 16);
}
__device__ __forceinline__ floatx8 fzero() {
    floatx8 v;
#pragma unroll
    for (int i = 0; i < 8; ++i) v[i] = 0.0f;
    return v;
}

// Build a v16 bf16 operand from two 16-byte chunks.
__device__ __forceinline__ bf16x16 ld2(const unsigned short* p0,
                                       const unsigned short* p1) {
    union { bf16x16 v; uint4 q[2]; } u;
    u.q[0] = *(const uint4*)p0;
    u.q[1] = *(const uint4*)p1;
    return u.v;
}

// A operand, 16x32 bf16 (ISA 7.12.2): lanes 0-15 row M=lane, K chunks {0..7,
// 16..23}; lanes 16-31 row M=lane-16, K chunks {8..15, 24..31}.
__device__ __forceinline__ bf16x16 load_a(const unsigned short* base, int ld,
                                          int row0, int k0, int lane) {
    const int r = row0 + (lane & 15);
    const int k = k0 + ((lane >> 4) << 3);          // 0 or 8
    const unsigned short* p = base + (size_t)r * ld + k;
    return ld2(p, p + 16);
}

// B operand, 32x16 bf16: lane n = column, K chunks matching A (lanes 0-15:
// K {0..7, 16..23}, lanes 16-31: K {8..15, 24..31}). `base` points at a matrix in
// which column n is a K-contiguous run with stride `ld` between columns.
__device__ __forceinline__ bf16x16 load_b(const unsigned short* base, int ld,
                                          int n0, int k0, int lane) {
    const int n = n0 + (lane & 15);
    const int k = k0 + ((lane >> 4) << 3);          // 0 or 8
    const unsigned short* p = base + (size_t)n * ld + k;
    return ld2(p, p + 16);
}

__device__ __forceinline__ floatx8 wmma_bf16(bf16x16 a, bf16x16 b, floatx8 c) {
    return __builtin_amdgcn_wmma_f32_16x16x32_bf16(
        false, a, false, b, (short)0, c, false, false);
}

// ----------------------------- prep kernels --------------------------------

__global__ void k_cvt(const float* __restrict__ s,
                      unsigned short* __restrict__ d, int n) {
    int i = blockIdx.x * blockDim.x + threadIdx.x;
    if (i < n) d[i] = f2b(s[i]);
}

// hB = bf16(concat(mem, x)) laid out [B][T][DM]
__global__ void k_cat(const float* __restrict__ mem,
                      const float* __restrict__ x,
                      unsigned short* __restrict__ hB) {
    size_t i = (size_t)blockIdx.x * blockDim.x + threadIdx.x;
    int d = (int)(i % DM);
    size_t row = i / DM;
    int t = (int)(row % TT);
    int b = (int)(row / TT);
    float v = (t < PREV) ? mem[((size_t)b * PREV + t) * DM + d]
                         : x[((size_t)b * CUR + (t - PREV)) * DM + d];
    hB[i] = f2b(v);
}

// vT[bh][d][t] = bf16 v-part of kv  (makes the attn@V B-operand K-contiguous)
__global__ void k_vtrans(const unsigned short* __restrict__ kvB,
                         unsigned short* __restrict__ vT) {
    size_t i = (size_t)blockIdx.x * blockDim.x + threadIdx.x;
    int t = (int)(i & (TT - 1));
    size_t r = i >> 11;
    int d = (int)(r & (DH - 1));
    int bh = (int)(r >> 6);
    int b = bh >> 4, h = bh & 15;
    vT[i] = kvB[((size_t)b * TT + t) * (2 * DM) + DM + h * DH + d];
}

// --------------------------- generic WMMA GEMM -----------------------------
// C[r][n] = sum_k A[r][k] * W[n][k]   (A row-major, W row-major = B col-major)
// 256 threads = 8 waves, each wave owns a 32x32 accumulator block; block tile
// is 64(M) x 128(N); K stepped by 32 (one WMMA depth).
// mode 0: o0 = bf16(acc)
// mode 1: o0 = bf16(acc + add0[n]); o1 = bf16(acc + add1[n])   (q+u / q+v)
// mode 2: of = acc + add0[r][n] + add1[n]                      (x + fc + b)
__global__ void k_gemm(const unsigned short* __restrict__ A, int lda,
                       const unsigned short* __restrict__ W, int ldw,
                       int K, int mode,
                       const float* __restrict__ add0,
                       const float* __restrict__ add1,
                       unsigned short* __restrict__ o0,
                       unsigned short* __restrict__ o1,
                       float* __restrict__ of, int ldc) {
    const int lane = threadIdx.x & 31;
    const int wave = threadIdx.x >> 5;
    const int row0 = blockIdx.x * 64 + (wave & 1) * 32;
    const int col0 = blockIdx.y * 128 + (wave >> 1) * 32;

    floatx8 acc[2][2];
    acc[0][0] = fzero(); acc[0][1] = fzero();
    acc[1][0] = fzero(); acc[1][1] = fzero();

    for (int k = 0; k < K; k += 32) {
        if (k + 64 < K) {   // keep next tiles warm in L2 (global_prefetch_b8)
            __builtin_prefetch(A + (size_t)row0 * lda + k + 64, 0, 1);
            __builtin_prefetch(W + (size_t)col0 * ldw + k + 64, 0, 1);
        }
        bf16x16 a0 = load_a(A, lda, row0,      k, lane);
        bf16x16 a1 = load_a(A, lda, row0 + 16, k, lane);
        bf16x16 b0 = load_b(W, ldw, col0,      k, lane);
        bf16x16 b1 = load_b(W, ldw, col0 + 16, k, lane);
        acc[0][0] = wmma_bf16(a0, b0, acc[0][0]);
        acc[0][1] = wmma_bf16(a0, b1, acc[0][1]);
        acc[1][0] = wmma_bf16(a1, b0, acc[1][0]);
        acc[1][1] = wmma_bf16(a1, b1, acc[1][1]);
    }

    const int mo = (lane >> 4) * 8;
    const int nn = lane & 15;
#pragma unroll
    for (int tm = 0; tm < 2; ++tm)
#pragma unroll
        for (int tn = 0; tn < 2; ++tn) {
#pragma unroll
            for (int g = 0; g < 8; ++g) {
                int r = row0 + tm * 16 + mo + g;
                int c = col0 + tn * 16 + nn;
                float val = acc[tm][tn][g];
                if (mode == 0) {
                    o0[(size_t)r * ldc + c] = f2b(val);
                } else if (mode == 1) {
                    o0[(size_t)r * ldc + c] = f2b(val + add0[c]);
                    o1[(size_t)r * ldc + c] = f2b(val + add1[c]);
                } else {
                    of[(size_t)r * ldc + c] =
                        val + add0[(size_t)r * ldc + c] + add1[c];
                }
            }
        }
}

// ------------------------------- scores ------------------------------------
// scores[bh][i][j] = bf16( mask ? ((qu_i . k_j) + (qv_src(i) . pe_j)) / 8
//                               : -inf )
// rel_shift folded into the qv row gather:  p = (b*I+i)+B ; src row valid iff
// p % (I+1) != 0.  Block = 128 threads (4 waves), each wave: 16(i) x 64(j).
__global__ void k_scores(const unsigned short* __restrict__ quB,
                         const unsigned short* __restrict__ qvB,
                         const unsigned short* __restrict__ kvB,
                         const unsigned short* __restrict__ peB,
                         const unsigned char* __restrict__ mask,
                         unsigned short* __restrict__ scores) {
    __shared__ alignas(16) unsigned short zbuf[16];
    if (threadIdx.x < 16) zbuf[threadIdx.x] = 0;
    __syncthreads();

    const int lane = threadIdx.x & 31;
    const int wave = threadIdx.x >> 5;
    const int bh = blockIdx.z;
    const int b = bh >> 4, h = bh & 15;
    const int i0 = blockIdx.y * 64 + wave * 16;
    const int j0 = blockIdx.x * 64;

    // per-lane A row addressing (content + rel-shifted pos)
    const int rl   = lane & 15;
    const int koff = (lane >> 4) << 3;                 // 0 or 8
    const int Rout = b * II + i0 + rl;
    const unsigned short* pa = quB + (size_t)Rout * DM + h * DH + koff;

    const int p  = Rout + BB;                          // rel_shift remap
    const int bp = p / (II + 1);
    const int ip = p - bp * (II + 1);
    const bool valid = (ip > 0);
    const int Rsrc = bp * II + ip - 1;
    const unsigned short* pq =
        valid ? (qvB + (size_t)Rsrc * DM + h * DH + koff) : zbuf;

    const unsigned short* kbase  = kvB + (size_t)b * TT * (2 * DM) + h * DH;
    const unsigned short* pebase = peB + h * DH;

    floatx8 acc[4];
#pragma unroll
    for (int jt = 0; jt < 4; ++jt) acc[jt] = fzero();

#pragma unroll
    for (int ks = 0; ks < DH; ks += 32) {
        bf16x16 a_c = ld2(pa + ks, pa + ks + 16);
        bf16x16 a_p = valid ? ld2(pq + ks, pq + ks + 16) : ld2(zbuf, zbuf);
#pragma unroll
        for (int jt = 0; jt < 4; ++jt) {
            bf16x16 b_c = load_b(kbase,  2 * DM, j0 + jt * 16, ks, lane);
            bf16x16 b_p = load_b(pebase, DM,     j0 + jt * 16, ks, lane);
            acc[jt] = wmma_bf16(a_c, b_c, acc[jt]);
            acc[jt] = wmma_bf16(a_p, b_p, acc[jt]);
        }
    }

    const int mo = (lane >> 4) * 8;
    const int nn = lane & 15;
#pragma unroll
    for (int jt = 0; jt < 4; ++jt) {
#pragma unroll
        for (int g = 0; g < 8; ++g) {
            int i = i0 + mo + g;
            int j = j0 + jt * 16 + nn;
            float s = acc[jt][g];
            unsigned char m = mask[((size_t)b * II + i) * JJ + j];
            s = m ? s * 0.125f : -INFINITY;      // mask before 1/sqrt(64)
            scores[(size_t)bh * II * JJ + (size_t)i * JJ + j] = f2b(s);
        }
    }
}

// ------------------- column softmax (over query axis i) --------------------
__global__ void k_softmax(unsigned short* __restrict__ scores) {
    const int bh = blockIdx.y;
    const int j = blockIdx.x * blockDim.x + threadIdx.x;
    unsigned short* base = scores + (size_t)bh * II * JJ + j;
    float m = -INFINITY;
    for (int i = 0; i < II; ++i) m = fmaxf(m, b2f(base[(size_t)i * JJ]));
    float sum = 0.0f;
    for (int i = 0; i < II; ++i) sum += __expf(b2f(base[(size_t)i * JJ]) - m);
    const float inv = 1.0f / sum;
    for (int i = 0; i < II; ++i) {
        float e = __expf(b2f(base[(size_t)i * JJ]) - m) * inv;
        base[(size_t)i * JJ] = f2b(e);
    }
}

// --------------------- weighted = attn @ V (per b,h) -----------------------
// Block = 128 threads (4 waves); wave: 16(i) x 64(d); K = J = 2048.
__global__ void k_weighted(const unsigned short* __restrict__ attn,
                           const unsigned short* __restrict__ vT,
                           unsigned short* __restrict__ wB) {
    const int lane = threadIdx.x & 31;
    const int wave = threadIdx.x >> 5;
    const int bh = blockIdx.y;
    const int b = bh >> 4, h = bh & 15;
    const int i0 = blockIdx.x * 64 + wave * 16;

    const unsigned short* Ab = attn + (size_t)bh * II * JJ;
    const unsigned short* Vb = vT + (size_t)bh * DH * TT;

    floatx8 acc[4];
#pragma unroll
    for (int dt = 0; dt < 4; ++dt) acc[dt] = fzero();

    for (int k = 0; k < JJ; k += 32) {
        bf16x16 a = load_a(Ab, JJ, i0, k, lane);
#pragma unroll
        for (int dt = 0; dt < 4; ++dt) {
            bf16x16 bm = load_b(Vb, TT, dt * 16, k, lane);
            acc[dt] = wmma_bf16(a, bm, acc[dt]);
        }
    }

    const int mo = (lane >> 4) * 8;
    const int nn = lane & 15;
#pragma unroll
    for (int dt = 0; dt < 4; ++dt)
#pragma unroll
        for (int g = 0; g < 8; ++g) {
            int i = i0 + mo + g;
            int d = dt * 16 + nn;
            wB[((size_t)b * II + i) * DM + h * DH + d] = f2b(acc[dt][g]);
        }
}

// ------------------------------ layernorm ----------------------------------
__global__ void k_ln(const float* __restrict__ yF,
                     const float* __restrict__ gamma,
                     const float* __restrict__ beta,
                     float* __restrict__ out) {
    __shared__ float s1[256], s2[256];
    const int r = blockIdx.x;
    const int tid = threadIdx.x;
    const float* y = yF + (size_t)r * DM;
    float a = 0.0f, q = 0.0f;
    for (int n = tid; n < DM; n += 256) { float v = y[n]; a += v; q += v * v; }
    s1[tid] = a; s2[tid] = q;
    __syncthreads();
    for (int st = 128; st > 0; st >>= 1) {
        if (tid < st) { s1[tid] += s1[tid + st]; s2[tid] += s2[tid + st]; }
        __syncthreads();
    }
    const float mu  = s1[0] * (1.0f / DM);
    const float var = s2[0] * (1.0f / DM) - mu * mu;
    const float rs  = rsqrtf(var + 1e-5f);
    for (int n = tid; n < DM; n += 256)
        out[(size_t)r * DM + n] = (y[n] - mu) * rs * gamma[n] + beta[n];
}

// ------------------------------- launcher ----------------------------------

extern "C" void kernel_launch(void* const* d_in, const int* in_sizes, int n_in,
                              void* d_out, int out_size, void* d_ws,
                              size_t ws_size, hipStream_t stream) {
    const float* x    = (const float*)d_in[0];
    const float* pos  = (const float*)d_in[1];
    const float* u    = (const float*)d_in[2];
    const float* v    = (const float*)d_in[3];
    const unsigned char* mask = (const unsigned char*)d_in[4]; // jnp.bool_
    const float* mem  = (const float*)d_in[5];
    const float* Wq   = (const float*)d_in[6];
    const float* Wkv  = (const float*)d_in[7];
    const float* Wfc  = (const float*)d_in[8];
    const float* bfc  = (const float*)d_in[9];
    const float* gamma= (const float*)d_in[10];
    const float* beta = (const float*)d_in[11];
    float* out = (float*)d_out;

    // workspace carve-out (~206 MB total)
    char* w = (char*)d_ws;
    size_t off = 0;
    auto carve = [&](size_t bytes) {
        void* p = w + off;
        off = (off + bytes + 255) & ~(size_t)255;
        return p;
    };
    unsigned short* hB   = (unsigned short*)carve((size_t)BB*TT*DM*2);   // 8 MB
    unsigned short* xB   = (unsigned short*)carve((size_t)BB*CUR*DM*2);  // 4 MB
    unsigned short* WqB  = (unsigned short*)carve((size_t)DM*DM*2);      // 2 MB
    unsigned short* WkvB = (unsigned short*)carve((size_t)2*DM*DM*2);    // 4 MB
    unsigned short* WfcB = (unsigned short*)carve((size_t)DM*DM*2);      // 2 MB
    unsigned short* peB  = (unsigned short*)carve((size_t)TT*DM*2);      // 4 MB
    unsigned short* quB  = (unsigned short*)carve((size_t)BB*CUR*DM*2);  // 4 MB
    unsigned short* qvB  = (unsigned short*)carve((size_t)BB*CUR*DM*2);  // 4 MB
    unsigned short* kvB  = (unsigned short*)carve((size_t)BB*TT*2*DM*2); // 16 MB
    unsigned short* vT   = (unsigned short*)carve((size_t)BB*NH*DH*TT*2);// 8 MB
    unsigned short* scB  = (unsigned short*)carve((size_t)BB*NH*II*JJ*2);// 128 MB
    unsigned short* wB   = (unsigned short*)carve((size_t)BB*CUR*DM*2);  // 4 MB
    float*          yF   = (float*)carve((size_t)BB*CUR*DM*4);           // 8 MB

    // 1) fp32 -> bf16 conversions + mem/x concat
    k_cvt<<<dim3(DM*DM/256), 256, 0, stream>>>(Wq,  WqB,  DM*DM);
    k_cvt<<<dim3(2*DM*DM/256), 256, 0, stream>>>(Wkv, WkvB, 2*DM*DM);
    k_cvt<<<dim3(DM*DM/256), 256, 0, stream>>>(Wfc, WfcB, DM*DM);
    k_cvt<<<dim3(TT*DM/256), 256, 0, stream>>>(pos, peB,  TT*DM);
    k_cvt<<<dim3(BB*CUR*DM/256), 256, 0, stream>>>(x, xB, BB*CUR*DM);
    k_cat<<<dim3(BB*TT*DM/256), 256, 0, stream>>>(mem, x, hB);

    // 2) q = x@Wq^T (+u / +v fused)            M=2048 N=1024 K=1024
    k_gemm<<<dim3(2048/64, 1024/128), 256, 0, stream>>>(
        xB, DM, WqB, DM, DM, 1, u, v, quB, qvB, nullptr, DM);

    // 3) kv = h@Wkv^T                          M=4096 N=2048 K=1024
    k_gemm<<<dim3(4096/64, 2048/128), 256, 0, stream>>>(
        hB, DM, WkvB, DM, DM, 0, nullptr, nullptr, kvB, nullptr, nullptr, 2*DM);

    // 4) scores = content + rel-shifted pos, mask, /8
    k_scores<<<dim3(JJ/64, II/64, BB*NH), 128, 0, stream>>>(
        quB, qvB, kvB, peB, mask, scB);

    // 5) softmax over the query axis (i), in place, bf16
    k_softmax<<<dim3(JJ/256, BB*NH), 256, 0, stream>>>(scB);

    // 6) transpose V part of kv -> vT[bh][d][t]
    k_vtrans<<<dim3(BB*NH*DH*TT/256), 256, 0, stream>>>(kvB, vT);

    // 7) weighted = attn @ V                   per (b,h): M=1024 N=64 K=2048
    k_weighted<<<dim3(II/64, BB*NH), 128, 0, stream>>>(scB, vT, wB);

    // 8) y = x + weighted@Wfc^T + bfc          M=2048 N=1024 K=1024
    k_gemm<<<dim3(2048/64, 1024/128), 256, 0, stream>>>(
        wB, DM, WfcB, DM, DM, 2, x, bfc, nullptr, nullptr, yF, DM);

    // 9) LayerNorm -> out
    k_ln<<<dim3(BB*CUR), 256, 0, stream>>>(yF, gamma, beta, out);
}
